// MultiHeadAttentionBase_65481071403464
// MI455X (gfx1250) — matrix operand order repaired and checked by reading, not compiled, not run on hardware
//
#include <hip/hip_runtime.h>
#include <hip/hip_bf16.h>
#include <math.h>

typedef __attribute__((ext_vector_type(16))) _Float16 v16h;
typedef __attribute__((ext_vector_type(8)))  _Float16 h8;
typedef __attribute__((ext_vector_type(4)))  _Float16 h4;
typedef __attribute__((ext_vector_type(2)))  __fp16   pk2;   // cvt_pkrtz result type
typedef __attribute__((ext_vector_type(8)))  float    v8f;
typedef __attribute__((ext_vector_type(4)))  float    f4;

#define B_   4
#define H_   16
#define S_   2048
#define DK_  64
#define WAVES 8
#define KV   64

// Padded LDS strides (halves): multiples of 8 halves (16B alignment for b128
// fragment chunks); 72*2B=144B stride gives conflict-free bank rotation.
#define QSTR 72
#define KSTR 72
#define VSTR 72
#define PSTR 72

#define LOG2E 1.4426950408889634f
#define MASKV (-1.0e9f * LOG2E)   // masked score, base-2 domain

__device__ inline v8f wmma_f16(v16h a, v16h b, v8f c) {
    return __builtin_amdgcn_wmma_f32_16x16x32_f16(
        false, a, false, b, (short)0, c, false, false);
}

// A-matrix 16x32 f16 fragment: row M = lane&15.
// lanes 0-15:  halves 0..7 -> K+0..7,  8..15 -> K+16..23
// lanes 16-31: halves 0..7 -> K+8..15, 8..15 -> K+24..31
__device__ inline v16h load_afrag(const _Float16* base, int stride, int lane, int kbase) {
    int row = lane & 15;
    int ko  = (lane >> 4) * 8;
    const _Float16* p = base + row * stride + kbase + ko;
    union { v16h v; h8 h[2]; } u;
    u.h[0] = *(const h8*)(p);
    u.h[1] = *(const h8*)(p + 16);
    return u.v;
}

// B-matrix 32x16 f16 fragment: column n = lane&15, K = (lane>>4)*8 + i (i<8), 16 + (lane>>4)*8 + (i-8) (i>=8).
// Storage base[n*stride + k] (contraction dim contiguous).
__device__ inline v16h load_bfrag(const _Float16* base, int stride, int lane,
                                  int nbase, int kbase) {
    int n  = nbase + (lane & 15);
    int k0 = kbase + ((lane >> 4) << 3);
    const _Float16* p = base + n * stride + k0;
    union { v16h v; h8 h[2]; } u;
    u.h[0] = *(const h8*)(p);
    u.h[1] = *(const h8*)(p + 16);
    return u.v;
}

// Butterfly reductions over the 16-lane row group using DPP row_ror (pure VALU,
// no DS traffic). Ctrl values must be literal constants: row_ror:8/4/2/1.
__device__ inline float rowmax16(float x) {
    int xi, yi;
    xi = __builtin_bit_cast(int, x);
    yi = __builtin_amdgcn_update_dpp(xi, xi, 0x128, 0xf, 0xf, false);
    x = fmaxf(x, __builtin_bit_cast(float, yi));
    xi = __builtin_bit_cast(int, x);
    yi = __builtin_amdgcn_update_dpp(xi, xi, 0x124, 0xf, 0xf, false);
    x = fmaxf(x, __builtin_bit_cast(float, yi));
    xi = __builtin_bit_cast(int, x);
    yi = __builtin_amdgcn_update_dpp(xi, xi, 0x122, 0xf, 0xf, false);
    x = fmaxf(x, __builtin_bit_cast(float, yi));
    xi = __builtin_bit_cast(int, x);
    yi = __builtin_amdgcn_update_dpp(xi, xi, 0x121, 0xf, 0xf, false);
    x = fmaxf(x, __builtin_bit_cast(float, yi));
    return x;
}
__device__ inline float rowsum16(float x) {
    int xi, yi;
    xi = __builtin_bit_cast(int, x);
    yi = __builtin_amdgcn_update_dpp(xi, xi, 0x128, 0xf, 0xf, false);
    x = x + __builtin_bit_cast(float, yi);
    xi = __builtin_bit_cast(int, x);
    yi = __builtin_amdgcn_update_dpp(xi, xi, 0x124, 0xf, 0xf, false);
    x = x + __builtin_bit_cast(float, yi);
    xi = __builtin_bit_cast(int, x);
    yi = __builtin_amdgcn_update_dpp(xi, xi, 0x122, 0xf, 0xf, false);
    x = x + __builtin_bit_cast(float, yi);
    xi = __builtin_bit_cast(int, x);
    yi = __builtin_amdgcn_update_dpp(xi, xi, 0x121, 0xf, 0xf, false);
    x = x + __builtin_bit_cast(float, yi);
    return x;
}

__device__ inline float fast_exp2(float x) {
    return __builtin_amdgcn_exp2f(x);   // bare v_exp_f32 (base-2)
}

__device__ inline h4 pack4(float a, float b, float c, float d) {
    union { h4 v; pk2 h[2]; } u;
    u.h[0] = __builtin_amdgcn_cvt_pkrtz(a, b);
    u.h[1] = __builtin_amdgcn_cvt_pkrtz(c, d);
    return u.v;
}

// ---- mask bit-packing: one u32 word = 32 columns, bit = (mask != 0) ----
__global__ __launch_bounds__(256)
void pack_mask_kernel(const int* __restrict__ mask, unsigned* __restrict__ packed) {
    size_t e = (size_t)blockIdx.x * 256 + threadIdx.x;
    int lane = threadIdx.x & 31;
    unsigned long long b = __ballot(mask[e] != 0);
    if (lane == 0) packed[e >> 5] = (unsigned)b;
}

template <bool PACKED>
__global__ __launch_bounds__(256)
void mha_flash_wmma(const float* __restrict__ q, const float* __restrict__ k,
                    const float* __restrict__ v, const int* __restrict__ mask,
                    const unsigned* __restrict__ mpk, float* __restrict__ out) {
    // Double-buffered K / V tiles; per-wave Q and P tiles. ~72 KB total.
    __shared__ __align__(16) _Float16 lds_k[2][KV * KSTR];       // [buf][kvrow][d]
    __shared__ __align__(16) _Float16 lds_vT[2][DK_ * VSTR];     // [buf][d][pos]
    __shared__ __align__(16) _Float16 lds_q[WAVES][16 * QSTR];   // [qrow][d]
    __shared__ __align__(16) _Float16 lds_p[WAVES][16 * PSTR];   // [qrow][pos]

    const int tid  = threadIdx.x;
    const int wave = tid >> 5;
    const int lane = tid & 31;

    const int qblocks = S_ / (WAVES * 16);          // 16
    const int bh   = blockIdx.x / qblocks;
    const int qblk = blockIdx.x % qblocks;
    const int b    = bh / H_;
    const size_t base = (size_t)bh * S_ * DK_;
    const int qbase = qblk * (WAVES * 16) + wave * 16;
    const float qscale = 0.125f * LOG2E;            // 1/sqrt(64) * log2(e), folded into Q

    const float* qp = q + base;
    const float* kp = k + base;
    const float* vp = v + base;
    const int*      mp  = mask + (size_t)b * S_ * S_;
    const unsigned* mpb = PACKED ? (mpk + (size_t)b * S_ * (S_ / 32)) : nullptr;
    float*       op = out + base;

    // K-staging thread mapping: 4 float4 slots per thread
    const int krow0 = tid >> 4;                      // reused: (i*256+tid)>>4 = i*16 + krow0
    const int kcol  = (tid & 15) << 2;
    // V-staging thread mapping (transpose group)
    const int vnc = tid >> 4;                        // kv column group 0..15
    const int vd4 = (tid & 15) << 2;                 // d base 0..60

    // ---- stage this wave's 16x64 Q tile (pre-scaled) into LDS as f16 ----
#pragma unroll
    for (int i = 0; i < 8; i++) {
        int e4  = i * 32 + lane;
        int row = e4 >> 4;
        int col = (e4 & 15) << 2;
        f4 g = *(const f4*)(qp + (size_t)(qbase + row) * DK_ + col);
        *(h4*)(&lds_q[wave][row * QSTR + col]) =
            pack4(g[0] * qscale, g[1] * qscale, g[2] * qscale, g[3] * qscale);
    }
    __builtin_amdgcn_wave_barrier();   // per-wave tile; DS in-order within wave
    v16h qa0 = load_afrag(&lds_q[wave][0], QSTR, lane, 0);
    v16h qa1 = load_afrag(&lds_q[wave][0], QSTR, lane, 32);

    v8f acc[4];
#pragma unroll
    for (int t = 0; t < 4; t++)
#pragma unroll
        for (int r = 0; r < 8; r++) acc[t][r] = 0.0f;
    float m[8], l[8];
#pragma unroll
    for (int r = 0; r < 8; r++) { m[r] = -INFINITY; l[r] = 0.0f; }

    const int rowbase = qbase + ((lane >> 4) << 3);
    const int ncol    = lane & 15;
    const int prow    = (lane >> 4) << 3;

    // tile-staging registers (next tile in flight while computing current)
    f4 kreg[4], vreg[4];

    // ---- load + store tile 0, one barrier ----
#pragma unroll
    for (int i = 0; i < 4; i++)
        kreg[i] = *(const f4*)(kp + (size_t)(i * 16 + krow0) * DK_ + kcol);
    vreg[0] = *(const f4*)(vp + (size_t)(vnc)      * DK_ + vd4);
    vreg[1] = *(const f4*)(vp + (size_t)(vnc + 16) * DK_ + vd4);
    vreg[2] = *(const f4*)(vp + (size_t)(vnc + 32) * DK_ + vd4);
    vreg[3] = *(const f4*)(vp + (size_t)(vnc + 48) * DK_ + vd4);
#pragma unroll
    for (int i = 0; i < 4; i++)
        *(h4*)(&lds_k[0][(i * 16 + krow0) * KSTR + kcol]) =
            pack4(kreg[i][0], kreg[i][1], kreg[i][2], kreg[i][3]);
#pragma unroll
    for (int dd = 0; dd < 4; dd++)
        *(h4*)(&lds_vT[0][(vd4 + dd) * VSTR + 4 * vnc]) =
            pack4(vreg[0][dd], vreg[1][dd], vreg[2][dd], vreg[3][dd]);
    __syncthreads();

    int cur = 0;
    for (int j = 0; j < S_; j += KV) {
        const bool more = (j + KV < S_);

        // ---- issue global loads for the NEXT tile (latency hidden by compute) ----
        if (more) {
            const float* kpn = kp + (size_t)(j + KV) * DK_;
            const float* vpn = vp + (size_t)(j + KV) * DK_;
#pragma unroll
            for (int i = 0; i < 4; i++)
                kreg[i] = *(const f4*)(kpn + (size_t)(i * 16 + krow0) * DK_ + kcol);
            vreg[0] = *(const f4*)(vpn + (size_t)(vnc)      * DK_ + vd4);
            vreg[1] = *(const f4*)(vpn + (size_t)(vnc + 16) * DK_ + vd4);
            vreg[2] = *(const f4*)(vpn + (size_t)(vnc + 32) * DK_ + vd4);
            vreg[3] = *(const f4*)(vpn + (size_t)(vnc + 48) * DK_ + vd4);
        }

        const _Float16* lk = &lds_k[cur][0];
        const _Float16* lv = &lds_vT[cur][0];

        // ---- S = Q*K^T : four 16x16 f32 fragments over 64 kv columns ----
        v8f s[4];
#pragma unroll
        for (int t = 0; t < 4; t++) {
#pragma unroll
            for (int r = 0; r < 8; r++) s[t][r] = 0.0f;
            s[t] = wmma_f16(qa0, load_bfrag(lk, KSTR, lane, t * 16, 0),  s[t]);
            s[t] = wmma_f16(qa1, load_bfrag(lk, KSTR, lane, t * 16, 32), s[t]);
        }

        // ---- mask + online softmax (base-2; scale pre-folded into Q) ----
        float alpha[8];
#pragma unroll
        for (int r = 0; r < 8; r++) {
            float sa = s[0][r], sb = s[1][r], sc = s[2][r], sd = s[3][r];
            if (PACKED) {
                const unsigned* wp = mpb + (size_t)(rowbase + r) * (S_ / 32) + (j >> 5);
                uint2 w = *(const uint2*)wp;
                unsigned a0 = w.x >> ncol;           // bit0 -> f0, bit16 -> f1
                unsigned a1 = w.y >> ncol;           // bit0 -> f2, bit16 -> f3
                sa = (a0 & 1u)       ? sa : MASKV;
                sb = (a0 & 0x10000u) ? sb : MASKV;
                sc = (a1 & 1u)       ? sc : MASKV;
                sd = (a1 & 0x10000u) ? sd : MASKV;
            } else {
                const int mrow = (rowbase + r) * S_ + j + ncol;
                if (mp[mrow]      == 0) sa = MASKV;
                if (mp[mrow + 16] == 0) sb = MASKV;
                if (mp[mrow + 32] == 0) sc = MASKV;
                if (mp[mrow + 48] == 0) sd = MASKV;
            }
            float rm = fmaxf(fmaxf(sa, sb), fmaxf(sc, sd));
            rm = rowmax16(rm);                       // DPP butterfly, no DS
            float mn = fmaxf(m[r], rm);
            alpha[r] = fast_exp2(m[r] - mn);
            m[r] = mn;
            float pa = fast_exp2(sa - mn);
            float pb = fast_exp2(sb - mn);
            float pc = fast_exp2(sc - mn);
            float pd = fast_exp2(sd - mn);
            // per-lane partial row sum; reduced across lanes once at the end
            l[r] = l[r] * alpha[r] + ((pa + pb) + (pc + pd));
            // P -> LDS, pos-permuted: pos = 4*ncol + f, packed b64 store
            *(h4*)(&lds_p[wave][(prow + r) * PSTR + 4 * ncol]) = pack4(pa, pb, pc, pd);
        }
        __builtin_amdgcn_wave_barrier();

        // rescale O accumulators by alpha (per row)
#pragma unroll
        for (int t = 0; t < 4; t++)
#pragma unroll
            for (int r = 0; r < 8; r++) acc[t][r] *= alpha[r];

        // ---- O += P * V  (k = pos index, 2 k-steps of 32) ----
        v16h pa0 = load_afrag(&lds_p[wave][0], PSTR, lane, 0);
        v16h pa1 = load_afrag(&lds_p[wave][0], PSTR, lane, 32);
#pragma unroll
        for (int t = 0; t < 4; t++) {
            acc[t] = wmma_f16(pa0, load_bfrag(lv, VSTR, lane, t * 16, 0),  acc[t]);
            acc[t] = wmma_f16(pa1, load_bfrag(lv, VSTR, lane, t * 16, 32), acc[t]);
        }

        // ---- convert + store the NEXT tile into the alternate buffer ----
        if (more) {
            const int nxt = cur ^ 1;
#pragma unroll
            for (int i = 0; i < 4; i++)
                *(h4*)(&lds_k[nxt][(i * 16 + krow0) * KSTR + kcol]) =
                    pack4(kreg[i][0], kreg[i][1], kreg[i][2], kreg[i][3]);
#pragma unroll
            for (int dd = 0; dd < 4; dd++)
                *(h4*)(&lds_vT[nxt][(vd4 + dd) * VSTR + 4 * vnc]) =
                    pack4(vreg[0][dd], vreg[1][dd], vreg[2][dd], vreg[3][dd]);
        }
        __syncthreads();   // single barrier per iteration
        cur ^= 1;
    }

    // ---- reduce per-lane partial l across the 16-lane row group, write out ----
#pragma unroll
    for (int r = 0; r < 8; r++) {
        float inv = 1.0f / rowsum16(l[r]);
#pragma unroll
        for (int t = 0; t < 4; t++) {
            op[(size_t)(rowbase + r) * DK_ + t * 16 + ncol] = acc[t][r] * inv;
        }
    }
}

extern "C" void kernel_launch(void* const* d_in, const int* in_sizes, int n_in,
                              void* d_out, int out_size, void* d_ws, size_t ws_size,
                              hipStream_t stream) {
    const float* q    = (const float*)d_in[0];
    const float* k    = (const float*)d_in[1];
    const float* v    = (const float*)d_in[2];
    const int*   mask = (const int*)d_in[3];
    float* out = (float*)d_out;

    const size_t packed_bytes = (size_t)B_ * S_ * (S_ / 32) * sizeof(unsigned);
    const bool use_packed = (ws_size >= packed_bytes);

    const int qblocks = S_ / (WAVES * 16);
    dim3 grid(B_ * H_ * qblocks);                // 1024 blocks
    dim3 block(WAVES * 32);                      // 256 threads = 8 wave32

    if (use_packed) {
        unsigned* mpk = (unsigned*)d_ws;
        const size_t nelem = (size_t)B_ * S_ * S_;
        pack_mask_kernel<<<dim3((unsigned)(nelem / 256)), dim3(256), 0, stream>>>(mask, mpk);
        mha_flash_wmma<true><<<grid, block, 0, stream>>>(q, k, v, mask, mpk, out);
    } else {
        mha_flash_wmma<false><<<grid, block, 0, stream>>>(q, k, v, mask, nullptr, out);
    }
}
